// ComplexAttention_30880814858909
// MI455X (gfx1250) — matrix operand order repaired and checked by reading, compiled, not run on hardware
//
#include <hip/hip_runtime.h>
#include <hip/hip_bf16.h>
#include <math.h>

// ---------------------------------------------------------------------------
// Complex multi-head attention for MI455X (gfx1250, wave32, WMMA + TDM).
// All contractions on v_wmma_f32_16x16x32_bf16; LDS tiles filled by the
// Tensor Data Mover (tensor_load_to_lds), double-buffered, s_wait_tensorcnt.
// ---------------------------------------------------------------------------

typedef __attribute__((ext_vector_type(16))) __bf16   v16bf;
typedef __attribute__((ext_vector_type(8)))  __bf16   v8bf;
typedef __attribute__((ext_vector_type(8)))  float    v8f;
typedef __attribute__((ext_vector_type(4)))  unsigned u32x4;
typedef __attribute__((ext_vector_type(8)))  unsigned u32x8;

namespace {

constexpr int kB = 2, kT = 1024, kC = 1024, kH = 16, kD = 64;
constexpr int kM = kB * kT;                         // 2048 token rows
constexpr float kRopeL2 = -0.20762050593046868f;    // -log2(10000)/64

// ---- WMMA fragment helpers (CDNA5 16-bit layouts, CDNA5 ISA) ----
// A (16xK=32): lane<16 -> row=lane, K chunks {0..7,16..23}; lane>=16 ->
// row=lane-16, K chunks {8..15,24..31}.  Two 16B loads from row-major src.
__device__ __forceinline__ v16bf frag_a(const __bf16* base, int ld, int row0,
                                        int k0, int lane) {
  const int hh = lane >> 4, li = lane & 15;
  const __bf16* p = base + (size_t)(row0 + li) * ld + k0 + hh * 8;
  v8bf c0 = *(const v8bf*)(p);
  v8bf c1 = *(const v8bf*)(p + 16);
  return __builtin_shufflevector(c0, c1, 0, 1, 2, 3, 4, 5, 6, 7,
                                 8, 9, 10, 11, 12, 13, 14, 15);
}

// B (K=32 x 16), source "output-feature row-major" (row n holds B[:,n] in k):
// lane<16 -> col=lane, K {0..7,16..23}; lane>=16 -> K {8..15,24..31}.
__device__ __forceinline__ v16bf frag_b(const __bf16* base, int ld, int col0,
                                        int k0, int lane) {
  const int hh = lane >> 4, li = lane & 15;
  const __bf16* p = base + (size_t)(col0 + li) * ld + k0 + hh * 8;
  v8bf c0 = *(const v8bf*)(p);
  v8bf c1 = *(const v8bf*)(p + 16);
  return __builtin_shufflevector(c0, c1, 0, 1, 2, 3, 4, 5, 6, 7,
                                 8, 9, 10, 11, 12, 13, 14, 15);
}

__device__ __forceinline__ v8f wmma_bf16(v16bf a, v16bf b, v8f c) {
  return __builtin_amdgcn_wmma_f32_16x16x32_bf16(
      false, a, false, b, (short)0, c, false, false);
}

__device__ __forceinline__ unsigned lds_off_of(const void* p) {
  // Generic LDS pointer = {SHARED_BASE, offset}; low 32 bits are the LDS addr.
  return (unsigned)(unsigned long long)p;
}

// ---- Tensor Data Mover: 2-D tile (bf16) global -> LDS ---------------------
// D# bitfields per CDNA5 ISA.  Groups 2/3 are
// zero-filled (both non-NULL -> 4-group descriptor; tile_dim3/4 = 0 unused).
__device__ __forceinline__ void tdm_load_2d(unsigned lds_off, const void* gp,
                                            unsigned tile_c,        // dim0 (contig) elems
                                            unsigned tile_r,        // dim1 rows
                                            unsigned stride_elems)  // row pitch
{
  const unsigned long long ga = (unsigned long long)gp;
  u32x4 g0;
  g0[0] = 1u;                                   // count=1, user descriptor
  g0[1] = lds_off;                              // lds_addr (bytes)
  g0[2] = (unsigned)ga;                         // global_addr[31:0]
  g0[3] = (unsigned)((ga >> 32) & 0x01ffffffull) | (2u << 30);  // [56:32]|type=2
  const unsigned TD = 1u << 20;                 // generous tensor extents (no OOB)
  u32x8 g1;
  g1[0] = 0x00010000u;                          // mask=0, data_size=2B
  g1[1] = (TD & 0xffffu) << 16;                 // tensor_dim0[15:0]
  g1[2] = (TD >> 16) | ((TD & 0xffffu) << 16);  // tensor_dim0[31:16] | dim1[15:0]
  g1[3] = (TD >> 16) | (tile_c << 16);          // tensor_dim1[31:16] | tile_dim0
  g1[4] = tile_r & 0xffffu;                     // tile_dim1 (tile_dim2 = 0)
  g1[5] = stride_elems;                         // tensor_dim0_stride[31:0]
  g1[6] = 0u;
  g1[7] = 0u;
  u32x4 gz = {0u, 0u, 0u, 0u};
  asm volatile("tensor_load_to_lds %0, %1, %2, %3"
               :: "s"(g0), "s"(g1), "s"(gz), "s"(gz)
               : "memory");
}

}  // namespace

// ---- Kernel 1: split interleaved complex f32 into bf16 re/im planes -------
__global__ __launch_bounds__(256) void pack_c2p(const float* __restrict__ src,
                                                __bf16* __restrict__ pr,
                                                __bf16* __restrict__ pi,
                                                int n) {
  int i = blockIdx.x * blockDim.x + threadIdx.x;
  if (i < n) {
    float2 v = ((const float2*)src)[i];
    pr[i] = (__bf16)v.x;
    pi[i] = (__bf16)v.y;
  }
}

// ---- Kernel 2: complex GEMM  y = a @ W^T + bias, TDM-staged ---------------
// Workgroup: 64(M) x 128(N) macro-tile, 8 waves as 2x4, each wave 2x2 tiles.
enum { MODE_ROPE = 0, MODE_VT = 1, MODE_OUT = 2 };

template <int MODE>
__global__ __launch_bounds__(256) void cgemm(
    const __bf16* __restrict__ ar, const __bf16* __restrict__ ai,
    const __bf16* __restrict__ wr, const __bf16* __restrict__ wi,
    const float* __restrict__ bias,
    __bf16* __restrict__ outr, __bf16* __restrict__ outi,
    float* __restrict__ outf) {
  __shared__ __attribute__((aligned(16))) __bf16 sA[2][2][64 * 32];   // 16 KB
  __shared__ __attribute__((aligned(16))) __bf16 sB[2][2][128 * 32];  // 32 KB

  const int lane = threadIdx.x & 31;
  const int wave = threadIdx.x >> 5;
  const int hh = lane >> 4, li = lane & 15;
  const int bm = blockIdx.x >> 3;   // 32 M-blocks of 64 rows
  const int bn = blockIdx.x & 7;    // 8 N-blocks of 128 cols
  const int row0 = bm * 64, col0 = bn * 128;
  const int wm = wave >> 2, wn = wave & 3;  // wave tile origin (32x32)

  v8f acc_rr[2][2] = {}, acc_ii[2][2] = {}, acc_ir[2][2] = {}, acc_ri[2][2] = {};

  constexpr int NSTEP = kC / 32;
  auto issue = [&](int step, int buf) {
    const int k0 = step * 32;
    tdm_load_2d(lds_off_of(&sA[buf][0][0]), ar + (size_t)row0 * kC + k0, 32, 64, kC);
    tdm_load_2d(lds_off_of(&sA[buf][1][0]), ai + (size_t)row0 * kC + k0, 32, 64, kC);
    tdm_load_2d(lds_off_of(&sB[buf][0][0]), wr + (size_t)col0 * kC + k0, 32, 128, kC);
    tdm_load_2d(lds_off_of(&sB[buf][1][0]), wi + (size_t)col0 * kC + k0, 32, 128, kC);
  };
  if (wave == 0) issue(0, 0);

  for (int step = 0; step < NSTEP; ++step) {
    const int buf = step & 1;
    __syncthreads();  // everyone done reading buf^1 from the previous step
    if (wave == 0) {
      if (step + 1 < NSTEP) {
        issue(step + 1, buf ^ 1);
        __builtin_amdgcn_s_wait_tensorcnt(4);  // in-order: step's 4 loads done
      } else {
        __builtin_amdgcn_s_wait_tensorcnt(0);
      }
    }
    __syncthreads();  // staged data for `buf` visible to all waves

    v16bf a_r[2], a_i[2], b_r[2], b_i[2];
#pragma unroll
    for (int i = 0; i < 2; ++i) {
      a_r[i] = frag_a(&sA[buf][0][0], 32, wm * 32 + i * 16, 0, lane);
      a_i[i] = frag_a(&sA[buf][1][0], 32, wm * 32 + i * 16, 0, lane);
    }
#pragma unroll
    for (int j = 0; j < 2; ++j) {
      b_r[j] = frag_b(&sB[buf][0][0], 32, wn * 32 + j * 16, 0, lane);
      b_i[j] = frag_b(&sB[buf][1][0], 32, wn * 32 + j * 16, 0, lane);
    }
#pragma unroll
    for (int i = 0; i < 2; ++i)
#pragma unroll
      for (int j = 0; j < 2; ++j) {
        acc_rr[i][j] = wmma_bf16(a_r[i], b_r[j], acc_rr[i][j]);
        acc_ii[i][j] = wmma_bf16(a_i[i], b_i[j], acc_ii[i][j]);
        acc_ir[i][j] = wmma_bf16(a_i[i], b_r[j], acc_ir[i][j]);
        acc_ri[i][j] = wmma_bf16(a_r[i], b_i[j], acc_ri[i][j]);
      }
  }

  // Epilogue: combine complex parts (bf16 WMMA NEG only supports CNeg).
#pragma unroll
  for (int i = 0; i < 2; ++i)
#pragma unroll
    for (int j = 0; j < 2; ++j) {
      const int col = col0 + wn * 32 + j * 16 + li;
      const float brv = bias[2 * col], biv = bias[2 * col + 1];
#pragma unroll
      for (int r = 0; r < 8; ++r) {
        const int row = row0 + wm * 32 + i * 16 + r + hh * 8;
        float yr = acc_rr[i][j][r] - acc_ii[i][j][r] + brv;
        float yi = acc_ir[i][j][r] + acc_ri[i][j][r] + biv;
        if constexpr (MODE == MODE_OUT) {
          size_t idx = ((size_t)row * kC + col) * 2;
          outf[idx] = yr;
          outf[idx + 1] = yi;
        } else {
          const int b = row >> 10, t = row & (kT - 1);
          const int h = col >> 6, d = col & (kD - 1);
          if constexpr (MODE == MODE_ROPE) {
            float ang = (float)t * exp2f((float)d * kRopeL2);
            float sn, cs;
            __sincosf(ang, &sn, &cs);
            float zr = yr * cs - yi * sn;
            float zi = yr * sn + yi * cs;
            size_t idx = ((size_t)(b * kH + h) * kT + t) * kD + d;  // [B,H,T,D]
            outr[idx] = (__bf16)zr;
            outi[idx] = (__bf16)zi;
          } else {  // MODE_VT: transposed so attn@V B-frags are K-contiguous
            size_t idx = ((size_t)(b * kH + h) * kD + d) * kT + t;  // [B,H,D,T]
            outr[idx] = (__bf16)yr;
            outi[idx] = (__bf16)yi;
          }
        }
      }
    }
}

// ---- Kernel 3: flash attention over |score|, complex V, TDM-staged K/V ----
// Block = one (b,h), 128 q-rows; 8 waves each own 16 rows.  K/V tiles for
// each 32-column step staged once per block (8x L2-traffic cut vs per-wave).
__global__ __launch_bounds__(256) void attn(
    const __bf16* __restrict__ qr, const __bf16* __restrict__ qi,
    const __bf16* __restrict__ kr, const __bf16* __restrict__ ki,
    const __bf16* __restrict__ vtr, const __bf16* __restrict__ vti,
    __bf16* __restrict__ outr, __bf16* __restrict__ outi) {
  __shared__ __attribute__((aligned(16))) __bf16 sK[2][2][32 * 64];  // 16 KB
  __shared__ __attribute__((aligned(16))) __bf16 sV[2][2][64 * 32];  // 16 KB
  __shared__ __attribute__((aligned(16))) __bf16 pst[8][16 * 32];    //  8 KB

  const int lane = threadIdx.x & 31;
  const int wave = threadIdx.x >> 5;
  const int hh = lane >> 4, li = lane & 15;
  const int bh = blockIdx.x >> 3;                       // (b*H + h)
  const int row0 = (blockIdx.x & 7) * 128 + wave * 16;  // q-row block in T

  const size_t pbase = (size_t)bh * kT * kD;
  const __bf16* q_r = qr + pbase;
  const __bf16* q_i = qi + pbase;
  const __bf16* k_r = kr + pbase;
  const __bf16* k_i = ki + pbase;
  const __bf16* v_r = vtr + pbase;  // [D,T]
  const __bf16* v_i = vti + pbase;
  __bf16* ps = &pst[wave][0];

  // q fragments (K = D = 64 -> two K=32 frags per component), loaded once.
  v16bf aqr0 = frag_a(q_r, kD, row0, 0, lane), aqr1 = frag_a(q_r, kD, row0, 32, lane);
  v16bf aqi0 = frag_a(q_i, kD, row0, 0, lane), aqi1 = frag_a(q_i, kD, row0, 32, lane);

  float mrow[8], lrow[8];
  v8f o_r[4] = {}, o_i[4] = {};
#pragma unroll
  for (int r = 0; r < 8; ++r) { mrow[r] = -1e30f; lrow[r] = 0.f; }

  auto issue = [&](int step, int buf) {
    const int s0 = step * 32;
    tdm_load_2d(lds_off_of(&sK[buf][0][0]), k_r + (size_t)s0 * kD, 64, 32, kD);
    tdm_load_2d(lds_off_of(&sK[buf][1][0]), k_i + (size_t)s0 * kD, 64, 32, kD);
    tdm_load_2d(lds_off_of(&sV[buf][0][0]), v_r + s0, 32, 64, kT);
    tdm_load_2d(lds_off_of(&sV[buf][1][0]), v_i + s0, 32, 64, kT);
  };
  if (wave == 0) issue(0, 0);

  constexpr int NSTEP = kT / 32;
  for (int step = 0; step < NSTEP; ++step) {
    const int buf = step & 1;
    __syncthreads();
    if (wave == 0) {
      if (step + 1 < NSTEP) {
        issue(step + 1, buf ^ 1);
        __builtin_amdgcn_s_wait_tensorcnt(4);
      } else {
        __builtin_amdgcn_s_wait_tensorcnt(0);
      }
    }
    __syncthreads();

    // score tiles: q * conj(k):  re = qr*kr + qi*ki ;  im = qi*kr - qr*ki
    float mag[2][8];
#pragma unroll
    for (int j = 0; j < 2; ++j) {
      v16bf bkr0 = frag_b(&sK[buf][0][0], 64, j * 16, 0, lane);
      v16bf bkr1 = frag_b(&sK[buf][0][0], 64, j * 16, 32, lane);
      v16bf bki0 = frag_b(&sK[buf][1][0], 64, j * 16, 0, lane);
      v16bf bki1 = frag_b(&sK[buf][1][0], 64, j * 16, 32, lane);
      v8f sre = {}, sim = {}, sng = {};
      sre = wmma_bf16(aqr0, bkr0, sre);
      sre = wmma_bf16(aqr1, bkr1, sre);
      sre = wmma_bf16(aqi0, bki0, sre);
      sre = wmma_bf16(aqi1, bki1, sre);
      sim = wmma_bf16(aqi0, bkr0, sim);
      sim = wmma_bf16(aqi1, bkr1, sim);
      sng = wmma_bf16(aqr0, bki0, sng);
      sng = wmma_bf16(aqr1, bki1, sng);
#pragma unroll
      for (int r = 0; r < 8; ++r) {
        float re = sre[r], im = sim[r] - sng[r];
        // Softmax magnitudes don't need IEEE-exact sqrt: raw v_sqrt_f32.
        mag[j][r] = __builtin_amdgcn_sqrtf(re * re + im * im) * 0.125f;
      }
    }
    // Online softmax: row = r + half*8, columns live in this half's 16 lanes.
#pragma unroll
    for (int r = 0; r < 8; ++r) {
      float pm = fmaxf(mag[0][r], mag[1][r]);
#pragma unroll
      for (int msk = 1; msk < 16; msk <<= 1) pm = fmaxf(pm, __shfl_xor(pm, msk, 32));
      const float mnew = fmaxf(mrow[r], pm);
      const float scale = __expf(mrow[r] - mnew);
      const float p0 = __expf(mag[0][r] - mnew);
      const float p1 = __expf(mag[1][r] - mnew);
      float rs = p0 + p1;
#pragma unroll
      for (int msk = 1; msk < 16; msk <<= 1) rs += __shfl_xor(rs, msk, 32);
      lrow[r] = lrow[r] * scale + rs;
      mrow[r] = mnew;
#pragma unroll
      for (int dt = 0; dt < 4; ++dt) { o_r[dt][r] *= scale; o_i[dt][r] *= scale; }
      const int prow = r + hh * 8;
      ps[prow * 32 + li] = (__bf16)p0;
      ps[prow * 32 + 16 + li] = (__bf16)p1;
    }
    // Intra-wave LDS RAW: drain DS stores before re-reading as A fragment.
    asm volatile("s_wait_dscnt 0" ::: "memory");
    v16bf ap = frag_a(ps, 32, 0, 0, lane);
    // attn @ V  (P real; V complex, staged transposed so K=s is contiguous)
#pragma unroll
    for (int dt = 0; dt < 4; ++dt) {
      v16bf bvr = frag_b(&sV[buf][0][0], 32, dt * 16, 0, lane);
      v16bf bvi = frag_b(&sV[buf][1][0], 32, dt * 16, 0, lane);
      o_r[dt] = wmma_bf16(ap, bvr, o_r[dt]);
      o_i[dt] = wmma_bf16(ap, bvi, o_i[dt]);
    }
  }

  // Normalize and store token-major bf16 planes [M, C] for final projection.
  const int b = bh >> 4, h = bh & 15;
#pragma unroll
  for (int r = 0; r < 8; ++r) {
    const float inv = __builtin_amdgcn_rcpf(lrow[r]);
    const int t = row0 + r + hh * 8;
    const size_t m = (size_t)b * kT + t;
#pragma unroll
    for (int dt = 0; dt < 4; ++dt) {
      const int c = h * kD + dt * 16 + li;
      outr[m * kC + c] = (__bf16)(o_r[dt][r] * inv);
      outi[m * kC + c] = (__bf16)(o_i[dt][r] * inv);
    }
  }
}

// ---------------------------------------------------------------------------
extern "C" void kernel_launch(void* const* d_in, const int* in_sizes, int n_in,
                              void* d_out, int out_size, void* d_ws,
                              size_t ws_size, hipStream_t stream) {
  (void)in_sizes; (void)n_in; (void)out_size; (void)ws_size;
  const float* x  = (const float*)d_in[0];
  const float* wq = (const float*)d_in[1];
  const float* bq = (const float*)d_in[2];
  const float* wk = (const float*)d_in[3];
  const float* bk = (const float*)d_in[4];
  const float* wv = (const float*)d_in[5];
  const float* bv = (const float*)d_in[6];
  const float* wo = (const float*)d_in[7];
  const float* bo = (const float*)d_in[8];
  float* out = (float*)d_out;

  char* ws = (char*)d_ws;
  size_t off = 0;
  auto take = [&](size_t bytes) -> __bf16* {
    char* p = ws + off;
    off += (bytes + 255) & ~(size_t)255;
    return (__bf16*)p;
  };
  __bf16* xr  = take((size_t)kM * kC * 2);
  __bf16* xi  = take((size_t)kM * kC * 2);
  __bf16* wqr = take((size_t)kC * kC * 2);
  __bf16* wqi = take((size_t)kC * kC * 2);
  __bf16* wkr = take((size_t)kC * kC * 2);
  __bf16* wki = take((size_t)kC * kC * 2);
  __bf16* wvr = take((size_t)kC * kC * 2);
  __bf16* wvi = take((size_t)kC * kC * 2);
  __bf16* wor = take((size_t)kC * kC * 2);
  __bf16* woi = take((size_t)kC * kC * 2);
  __bf16* q_r = take((size_t)kM * kC * 2);   // [B,H,T,D]
  __bf16* q_i = take((size_t)kM * kC * 2);
  __bf16* k_r = take((size_t)kM * kC * 2);
  __bf16* k_i = take((size_t)kM * kC * 2);
  __bf16* vt_r = take((size_t)kM * kC * 2);  // [B,H,D,T]
  __bf16* vt_i = take((size_t)kM * kC * 2);
  __bf16* o_r = take((size_t)kM * kC * 2);   // [M,C]
  __bf16* o_i = take((size_t)kM * kC * 2);

  const dim3 blk(256);
  const int nx = kM * kC, nw = kC * kC;
  pack_c2p<<<(nx + 255) / 256, blk, 0, stream>>>(x, xr, xi, nx);
  pack_c2p<<<(nw + 255) / 256, blk, 0, stream>>>(wq, wqr, wqi, nw);
  pack_c2p<<<(nw + 255) / 256, blk, 0, stream>>>(wk, wkr, wki, nw);
  pack_c2p<<<(nw + 255) / 256, blk, 0, stream>>>(wv, wvr, wvi, nw);
  pack_c2p<<<(nw + 255) / 256, blk, 0, stream>>>(wo, wor, woi, nw);

  const int gemm_blocks = (kM / 64) * (kC / 128);  // 256 macro-tiles
  cgemm<MODE_ROPE><<<gemm_blocks, blk, 0, stream>>>(xr, xi, wqr, wqi, bq,
                                                    q_r, q_i, nullptr);
  cgemm<MODE_ROPE><<<gemm_blocks, blk, 0, stream>>>(xr, xi, wkr, wki, bk,
                                                    k_r, k_i, nullptr);
  cgemm<MODE_VT><<<gemm_blocks, blk, 0, stream>>>(xr, xi, wvr, wvi, bv,
                                                  vt_r, vt_i, nullptr);

  const int attn_blocks = kB * kH * (kT / 128);  // 256
  attn<<<attn_blocks, blk, 0, stream>>>(q_r, q_i, k_r, k_i, vt_r, vt_i,
                                        o_r, o_i);

  cgemm<MODE_OUT><<<gemm_blocks, blk, 0, stream>>>(o_r, o_i, wor, woi, bo,
                                                   nullptr, nullptr, out);
}
